// BackwardDiagMVN_72241349919192
// MI455X (gfx1250) — matrix operand order repaired and checked by reading, not compiled, not run on hardware
//
#include <hip/hip_runtime.h>
#include <hip/hip_bf16.h>
#include <math.h>

// Problem sizes (fixed by the reference)
#define T_LEN 32768
#define D_IN  512
#define H_DIM 1024
#define S_DIM 256
#define NG3H  (3 * H_DIM)   // 3072
#define NOUT  (2 * S_DIM)   // 512

// Persistent scan configuration
#define NWG   128
#define HP    (H_DIM / NWG)   // 8 h-elements per workgroup
#define ROWS  (3 * HP)        // 24 w_hh rows per workgroup

typedef __attribute__((ext_vector_type(16))) __bf16    v16bf;
typedef __attribute__((ext_vector_type(8)))  float     v8f;
typedef __attribute__((ext_vector_type(4)))  unsigned  v4u;
typedef __attribute__((ext_vector_type(8)))  int       v8i;
typedef __attribute__((ext_vector_type(4)))  int       v4i;

#if defined(__has_builtin)
#  if __has_builtin(__builtin_amdgcn_tensor_load_to_lds) && \
      __has_builtin(__builtin_amdgcn_s_wait_tensorcnt)
#    define USE_TDM 1
#  endif
#endif
#ifndef USE_TDM
#  define USE_TDM 0
#endif

// ---------------------------------------------------------------------------
// Fragment builders for v_wmma_f32_16x16x32_bf16 (contiguous k order chosen here, same for A and B):
//   A (16x32, 16-bit): lanes 0-15 (row m) hold K = 0..15;
//                      lanes 16-31 hold K = 16..31
//   B (32x16, 16-bit): lanes 0-15 (col n) hold K = 0..15; lanes 16-31 K = 16..31
// Both map to contiguous float4 global loads; fp32 -> bf16 converted in-register.
// ---------------------------------------------------------------------------
__device__ __forceinline__ v16bf load_frag_a(const float* __restrict__ X,
                                             size_t row_base, int kb, int half) {
  const int ka = kb + half * 16;
  const float4* p0 = reinterpret_cast<const float4*>(X + row_base + ka);
  const float4* p1 = reinterpret_cast<const float4*>(X + row_base + ka + 8);
  float4 a0 = p0[0], a1 = p0[1], a2 = p1[0], a3 = p1[1];
  v16bf f;
  f[0]  = (__bf16)a0.x; f[1]  = (__bf16)a0.y; f[2]  = (__bf16)a0.z; f[3]  = (__bf16)a0.w;
  f[4]  = (__bf16)a1.x; f[5]  = (__bf16)a1.y; f[6]  = (__bf16)a1.z; f[7]  = (__bf16)a1.w;
  f[8]  = (__bf16)a2.x; f[9]  = (__bf16)a2.y; f[10] = (__bf16)a2.z; f[11] = (__bf16)a2.w;
  f[12] = (__bf16)a3.x; f[13] = (__bf16)a3.y; f[14] = (__bf16)a3.z; f[15] = (__bf16)a3.w;
  return f;
}

__device__ __forceinline__ v16bf load_frag_b(const float* __restrict__ W,
                                             size_t col_base, int kb, int half) {
  const int kc = kb + half * 16;
  const float4* p = reinterpret_cast<const float4*>(W + col_base + kc);
  float4 b0 = p[0], b1 = p[1], b2 = p[2], b3 = p[3];
  v16bf f;
  f[0]  = (__bf16)b0.x; f[1]  = (__bf16)b0.y; f[2]  = (__bf16)b0.z; f[3]  = (__bf16)b0.w;
  f[4]  = (__bf16)b1.x; f[5]  = (__bf16)b1.y; f[6]  = (__bf16)b1.z; f[7]  = (__bf16)b1.w;
  f[8]  = (__bf16)b2.x; f[9]  = (__bf16)b2.y; f[10] = (__bf16)b2.z; f[11] = (__bf16)b2.w;
  f[12] = (__bf16)b3.x; f[13] = (__bf16)b3.y; f[14] = (__bf16)b3.z; f[15] = (__bf16)b3.w;
  return f;
}

// One wave computes a 16(M) x 64(N) strip: 4 accumulators sharing one A fragment.
__device__ __forceinline__ void wmma_strip4(const float* __restrict__ X,
                                            const float* __restrict__ W,
                                            int m0, int n0, int K,
                                            int ldx, int ldw, int lane,
                                            v8f acc[4]) {
  const int sub  = lane & 15;
  const int half = lane >> 4;
  const size_t arow = (size_t)(m0 + sub) * ldx;
  size_t brow[4];
#pragma unroll
  for (int j = 0; j < 4; ++j) brow[j] = (size_t)(n0 + j * 16 + sub) * ldw;

  for (int kb = 0; kb < K; kb += 32) {
    const v16bf af = load_frag_a(X, arow, kb, half);
#pragma unroll
    for (int j = 0; j < 4; ++j) {
      const v16bf bf_ = load_frag_b(W, brow[j], kb, half);
      acc[j] = __builtin_amdgcn_wmma_f32_16x16x32_bf16(false, af, false, bf_,
                                                       (short)0, acc[j], false, false);
    }
  }
}

// ---------------------------------------------------------------------------
// Kernel A: igates[T,3H] = y[T,D] @ w_ih[3H,D]^T + b
// 256 threads = 8 waves; each wave computes a 16x64 strip; block = 16M x 512N.
// ---------------------------------------------------------------------------
__global__ void __launch_bounds__(256)
igates_kernel(const float* __restrict__ y, const float* __restrict__ w_ih,
              const float* __restrict__ b, float* __restrict__ igates) {
  const int lane = threadIdx.x & 31;
  const int wave = threadIdx.x >> 5;
  const int m0 = blockIdx.x * 16;
  const int n0 = blockIdx.y * 512 + wave * 64;

  v8f acc[4] = {{0.f,0.f,0.f,0.f,0.f,0.f,0.f,0.f},
                {0.f,0.f,0.f,0.f,0.f,0.f,0.f,0.f},
                {0.f,0.f,0.f,0.f,0.f,0.f,0.f,0.f},
                {0.f,0.f,0.f,0.f,0.f,0.f,0.f,0.f}};
  wmma_strip4(y, w_ih, m0, n0, D_IN, D_IN, D_IN, lane, acc);

  const int half = lane >> 4;
  const int sub  = lane & 15;
#pragma unroll
  for (int j = 0; j < 4; ++j) {
    const int n = n0 + j * 16 + sub;
    const float bias = b[n];
#pragma unroll
    for (int v = 0; v < 8; ++v) {
      const int m = m0 + half * 8 + v;  // C layout: lanes<16 -> rows 0..7, lanes>=16 -> 8..15
      igates[(size_t)m * NG3H + n] = acc[j][v] + bias;
    }
  }
}

// ---------------------------------------------------------------------------
// Init: zero the grid-sync counter (must happen every launch -> graph-safe).
// ---------------------------------------------------------------------------
__global__ void init_cnt_kernel(unsigned* cnt) {
  if (threadIdx.x == 0) *cnt = 0u;
}

// ---------------------------------------------------------------------------
// Kernel B: persistent GRU scan. 128 WGs x 256 threads. WG g owns h-elements
// [g*8, g*8+8) and keeps its 24 rows of w_hh (r/z/n slices) in LDS as bf16.
// Per step: 3 dot products per wave, gate math on lane 0, publish 8 elements
// to double-buffered global h, grid barrier, TDM-broadcast h back into LDS.
// ---------------------------------------------------------------------------
__global__ void __launch_bounds__(256)
gru_scan_kernel(const float* __restrict__ h0, const float* __restrict__ w_hh,
                const float* __restrict__ bn, const float* __restrict__ igates,
                float* __restrict__ hs, float* __restrict__ h_buf,
                unsigned* __restrict__ cnt) {
  __shared__ __bf16 w_lds[ROWS][H_DIM];   // 48 KB
  __shared__ float  h_lds[H_DIM];         // 4 KB

  const int g    = blockIdx.x;
  const int tid  = threadIdx.x;
  const int lane = tid & 31;
  const int wave = tid >> 5;

  // Stage this WG's 24 rows of w_hh (gate-major: r rows, z rows, n rows).
  for (int idx = tid; idx < ROWS * H_DIM; idx += 256) {
    const int r = idx >> 10;          // local row 0..23
    const int k = idx & (H_DIM - 1);
    const int gate = r / HP;
    const int i    = r - gate * HP;
    const int grow = gate * H_DIM + g * HP + i;
    w_lds[r][k] = (__bf16)w_hh[(size_t)grow * H_DIM + k];
  }
  // h_{-1} = tanh(h0), private copy per WG.
  for (int k = tid; k < H_DIM; k += 256) h_lds[k] = tanhf(h0[k]);
  __syncthreads();

  const int iG = g * HP + wave;        // the single h-element this wave owns
  const float bn_i = bn[iG];
  const __bf16* wr = w_lds[wave];
  const __bf16* wz = w_lds[HP + wave];
  const __bf16* wn = w_lds[2 * HP + wave];

#pragma unroll 1
  for (int t = 0; t < T_LEN; ++t) {
    // Input-side gates for this step (lane 0 only; prefetch next step).
    float ir = 0.f, iz = 0.f, inn = 0.f;
    if (lane == 0) {
      const float* ig = igates + (size_t)t * NG3H;
      ir  = ig[iG];
      iz  = ig[H_DIM + iG];
      inn = ig[2 * H_DIM + iG];
      if (t + 1 < T_LEN)
        __builtin_prefetch(igates + (size_t)(t + 1) * NG3H + iG, 0, 0);
    }
    // Three 1024-length dot products, 32-way split across the wave.
    float pr = 0.f, pz = 0.f, pn = 0.f;
#pragma unroll 8
    for (int j = 0; j < H_DIM / 32; ++j) {
      const int k = j * 32 + lane;
      const float hv = h_lds[k];
      pr = fmaf((float)wr[k], hv, pr);
      pz = fmaf((float)wz[k], hv, pz);
      pn = fmaf((float)wn[k], hv, pn);
    }
#pragma unroll
    for (int off = 16; off; off >>= 1) {
      pr += __shfl_down(pr, off, 32);
      pz += __shfl_down(pz, off, 32);
      pn += __shfl_down(pn, off, 32);
    }
    const int par = t & 1;
    if (lane == 0) {
      const float r = 1.f / (1.f + __expf(-(ir + pr)));
      const float z = 1.f / (1.f + __expf(-(iz + pz)));
      const float n = tanhf(inn + r * (pn + bn_i));
      const float hnew = n + z * (h_lds[iG] - n);
      h_buf[par * H_DIM + iG] = hnew;
      hs[(size_t)t * H_DIM + iG] = hnew;
    }
    __threadfence();
    __syncthreads();
    // Grid barrier: monotonic counter, target (t+1)*NWG.
    if (tid == 0) {
      __hip_atomic_fetch_add(cnt, 1u, __ATOMIC_ACQ_REL, __HIP_MEMORY_SCOPE_AGENT);
      const unsigned target = (unsigned)(t + 1) * (unsigned)NWG;
      while (__hip_atomic_load(cnt, __ATOMIC_ACQUIRE, __HIP_MEMORY_SCOPE_AGENT) < target)
        __builtin_amdgcn_s_sleep(2);
    }
    __syncthreads();

    // Pull the full updated h into LDS for the next step.
#if USE_TDM
    // One Tensor-Data-Mover transfer: 1024x1 tile of 4-byte elements,
    // global h_buf[par] -> LDS h_lds.  D# packed per ISA 08_async_tensor.
    // This toolchain exposes the 6-arg builtin:
    //   (uint32x4 g0, int32x8 g1, int32x4 g2, int32x4 g3, int32x8 g4, i32 cpol)
    if (tid < 32) {
      const unsigned lds_off = (unsigned)(uintptr_t)&h_lds[0];
      const unsigned long long ga =
          (unsigned long long)(uintptr_t)(h_buf + (size_t)par * H_DIM);
      v4u g0 = {0u, 0u, 0u, 0u};
      g0[0] = 1u;                                    // count=1, user mode
      g0[1] = lds_off;                               // lds_addr
      g0[2] = (unsigned)(ga & 0xffffffffu);          // global_addr[31:0]
      g0[3] = (unsigned)((ga >> 32) & 0x01ffffffu)   // global_addr[56:32]
              | (2u << 30);                          // type = 2 (image)
      v8i g1 = {0, 0, 0, 0, 0, 0, 0, 0};
      g1[0] = (int)(2u << 16);                       // data_size = 4 bytes
      g1[1] = (int)(((unsigned)H_DIM & 0xffffu) << 16); // tensor_dim0 lo16
      g1[2] = (int)(1u << 16);                       // tensor_dim1 = 1
      g1[3] = (int)(((unsigned)H_DIM & 0xffffu) << 16); // tile_dim0 = 1024
      g1[4] = 1;                                     // tile_dim1 = 1
      g1[5] = H_DIM;                                 // tensor_dim0_stride
      v4i g2 = {0, 0, 0, 0};
      v4i g3 = {0, 0, 0, 0};
      v8i g4 = {0, 0, 0, 0, 0, 0, 0, 0};
      __builtin_amdgcn_tensor_load_to_lds(g0, g1, g2, g3, g4, 0);
      __builtin_amdgcn_s_wait_tensorcnt(0);
    }
    __syncthreads();
#else
    for (int k = tid; k < H_DIM; k += 256) h_lds[k] = h_buf[par * H_DIM + k];
    __syncthreads();
#endif
  }
}

// ---------------------------------------------------------------------------
// Kernel C: out = hs[T,H] @ w_out[2S,H]^T + b_out; first S cols -> nat1,
// last S cols -> softplus -> nat2 (outputs concatenated flat in d_out).
// ---------------------------------------------------------------------------
__global__ void __launch_bounds__(256)
out_proj_kernel(const float* __restrict__ hs, const float* __restrict__ w_out,
                const float* __restrict__ b_out, float* __restrict__ out) {
  const int lane = threadIdx.x & 31;
  const int wave = threadIdx.x >> 5;
  const int m0 = blockIdx.x * 16;
  const int n0 = wave * 64;           // 8 waves x 64 = 512 = full N

  v8f acc[4] = {{0.f,0.f,0.f,0.f,0.f,0.f,0.f,0.f},
                {0.f,0.f,0.f,0.f,0.f,0.f,0.f,0.f},
                {0.f,0.f,0.f,0.f,0.f,0.f,0.f,0.f},
                {0.f,0.f,0.f,0.f,0.f,0.f,0.f,0.f}};
  wmma_strip4(hs, w_out, m0, n0, H_DIM, H_DIM, H_DIM, lane, acc);

  const int half = lane >> 4;
  const int sub  = lane & 15;
#pragma unroll
  for (int j = 0; j < 4; ++j) {
    const int n = n0 + j * 16 + sub;
    const float bias = b_out[n];
#pragma unroll
    for (int v = 0; v < 8; ++v) {
      const int t = m0 + half * 8 + v;
      const float val = acc[j][v] + bias;
      if (n < S_DIM) {
        out[(size_t)t * S_DIM + n] = val;                      // nat1
      } else {
        const float sp = fmaxf(val, 0.f) + log1pf(__expf(-fabsf(val)));
        out[(size_t)T_LEN * S_DIM + (size_t)t * S_DIM + (n - S_DIM)] = sp;  // nat2
      }
    }
  }
}

// ---------------------------------------------------------------------------
extern "C" void kernel_launch(void* const* d_in, const int* in_sizes, int n_in,
                              void* d_out, int out_size, void* d_ws, size_t ws_size,
                              hipStream_t stream) {
  (void)in_sizes; (void)n_in; (void)out_size; (void)ws_size;
  const float* y     = (const float*)d_in[0];
  const float* h0    = (const float*)d_in[1];
  const float* w_ih  = (const float*)d_in[2];
  const float* w_hh  = (const float*)d_in[3];
  const float* b     = (const float*)d_in[4];
  const float* bn    = (const float*)d_in[5];
  const float* w_out = (const float*)d_in[6];
  const float* b_out = (const float*)d_in[7];
  float* out = (float*)d_out;

  // Workspace layout
  float*    igates = (float*)d_ws;                            // T*3H
  float*    hs     = igates + (size_t)T_LEN * NG3H;           // T*H
  float*    h_buf  = hs + (size_t)T_LEN * H_DIM;              // 2*H (double buffer)
  unsigned* cnt    = (unsigned*)(h_buf + 2 * H_DIM);          // grid-sync counter

  // 1) input-side gates GEMM (WMMA, 16x64 strip per wave)
  igates_kernel<<<dim3(T_LEN / 16, NG3H / 512), 256, 0, stream>>>(y, w_ih, b, igates);
  // 2) reset sync counter (every launch -> graph replay safe)
  init_cnt_kernel<<<1, 32, 0, stream>>>(cnt);
  // 3) persistent sequential GRU scan (TDM h-broadcast per step)
  gru_scan_kernel<<<NWG, 256, 0, stream>>>(h0, w_hh, bn, igates, hs, h_buf, cnt);
  // 4) output projection GEMM + softplus split (WMMA)
  out_proj_kernel<<<dim3(T_LEN / 16, 1), 256, 0, stream>>>(hs, w_out, b_out, out);
}
